// MoELayer_90202903150800
// MI455X (gfx1250) — compile-verified, matrix operand order checked by reading, not run on hardware
//
#include <hip/hip_runtime.h>
#include <math.h>

// ---------------------------------------------------------------------------
// MoE forward for MI455X (gfx1250, wave32, WMMA).
// Routed top-2 implementation: ~275 GFLOP vs the reference's dense 1.1 TFLOP.
// Core GEMMs use v_wmma_f32_16x16x32_bf16 (full-rate CDNA5 matrix path),
// f32 accumulate. Workspace use: ~75.7 MB (bf16 H activation buffer).
// ---------------------------------------------------------------------------

typedef __bf16 bf16_t;
typedef __attribute__((ext_vector_type(16))) __bf16 v16bf;
typedef __attribute__((ext_vector_type(8)))  float  v8f;

#define D_MODEL 1024
#define D_FF    4096
#define NEXP    8
#define TOPK    2
#define NTOK    4096                 // BATCH * SEQ
#define NSLOT_CAP 9216               // 8192 real slots + per-expert pad to 128

#define TM 128                       // token tile (M)
#define TN 128                       // output-col tile (N)
#define TK 32                        // k step == WMMA K
#define LSTR 40                      // padded LDS row stride in bf16 (80 B, 16B-aligned)

// workspace byte offsets
#define OFF_CNT   0                                   // 8 int
#define OFF_OFFS  64                                  // 9 int (padded prefix)
#define OFF_CUR   128                                 // 8 int (scatter cursors)
#define OFF_TOPI  256                                 // NTOK*2 int
#define OFF_TOPW  (OFF_TOPI + NTOK*TOPK*4)            // NTOK*2 float
#define OFF_SLOTT (OFF_TOPW + NTOK*TOPK*4)            // NSLOT_CAP int
#define OFF_SLOTW (OFF_SLOTT + NSLOT_CAP*4)           // NSLOT_CAP float
#define OFF_H     ((OFF_SLOTW + NSLOT_CAP*4 + 255) & ~255)  // NSLOT_CAP*D_FF bf16

__device__ __forceinline__ float gelu_exact(float v) {
    return 0.5f * v * (1.0f + erff(v * 0.70710678118654752440f));
}

// ---------------------------------------------------------------------------
// 0) init: zero d_out (harness poisons it), counters, slot tables
// ---------------------------------------------------------------------------
__global__ void moe_init(float* __restrict__ out, int* __restrict__ cnt,
                         int* __restrict__ slotT, float* __restrict__ slotW) {
    size_t g = (size_t)blockIdx.x * blockDim.x + threadIdx.x;
    size_t stride = (size_t)gridDim.x * blockDim.x;
    for (size_t i = g; i < (size_t)NTOK * D_MODEL; i += stride) out[i] = 0.0f;
    if (g < NSLOT_CAP) { slotT[g] = -1; slotW[g] = 0.0f; }
    if (g < NEXP) cnt[g] = 0;
}

// ---------------------------------------------------------------------------
// 1) router: one wave32 per token
// ---------------------------------------------------------------------------
__global__ __launch_bounds__(256) void moe_router(
    const float* __restrict__ x, const float* __restrict__ Wr,
    const float* __restrict__ br, int* __restrict__ cnt,
    int* __restrict__ topI, float* __restrict__ topW) {
    const int lane = threadIdx.x & 31;
    const int t = blockIdx.x * 8 + (threadIdx.x >> 5);

    float acc[NEXP];
#pragma unroll
    for (int e = 0; e < NEXP; ++e) acc[e] = 0.0f;

    const float* xr = x + (size_t)t * D_MODEL;
    for (int k = lane; k < D_MODEL; k += 32) {
        const float xv = xr[k];
#pragma unroll
        for (int e = 0; e < NEXP; ++e) acc[e] += xv * Wr[k * NEXP + e];
    }
#pragma unroll
    for (int e = 0; e < NEXP; ++e) {
#pragma unroll
        for (int off = 16; off > 0; off >>= 1)
            acc[e] += __shfl_xor(acc[e], off, 32);
    }
    if (lane == 0) {
        float l[NEXP];
#pragma unroll
        for (int e = 0; e < NEXP; ++e) l[e] = acc[e] + br[e];
        int i0 = 0;
#pragma unroll
        for (int e = 1; e < NEXP; ++e) if (l[e] > l[i0]) i0 = e;
        int i1 = (i0 == 0) ? 1 : 0;
#pragma unroll
        for (int e = 0; e < NEXP; ++e) if (e != i0 && l[e] > l[i1]) i1 = e;
        // softmax + top-2 renorm collapses to a 2-way logistic (stable: l1<=l0)
        const float w0 = 1.0f / (1.0f + expf(l[i1] - l[i0]));
        topI[t * 2 + 0] = i0;  topI[t * 2 + 1] = i1;
        topW[t * 2 + 0] = w0;  topW[t * 2 + 1] = 1.0f - w0;
        atomicAdd(&cnt[i0], 1);
        atomicAdd(&cnt[i1], 1);
    }
}

// ---------------------------------------------------------------------------
// 2) plan: padded prefix offsets, cursors, aux load-balance loss cv^2
// ---------------------------------------------------------------------------
__global__ void moe_plan(const int* __restrict__ cnt, int* __restrict__ offs,
                         int* __restrict__ cur, float* __restrict__ out_cv2) {
    if (threadIdx.x != 0) return;
    int off = 0;
    for (int e = 0; e < NEXP; ++e) {
        offs[e] = off;
        cur[e] = off;
        off += ((cnt[e] + TM - 1) / TM) * TM;   // pad to M-tile
    }
    offs[NEXP] = off;

    float util[NEXP], s = 0.0f;
    for (int e = 0; e < NEXP; ++e) { util[e] = (float)cnt[e] / (float)(NTOK * TOPK); s += util[e]; }
    const float mean = s / (float)NEXP;
    float var = 0.0f;
    for (int e = 0; e < NEXP; ++e) { const float d = util[e] - mean; var += d * d; }
    var /= (float)(NEXP - 1);                    // ddof=1
    const float cv = sqrtf(var) / (mean + 1e-6f);
    *out_cv2 = cv * cv;
}

// ---------------------------------------------------------------------------
// 3) scatter tokens into per-expert slot lists
// ---------------------------------------------------------------------------
__global__ void moe_scatter(const int* __restrict__ topI, const float* __restrict__ topW,
                            int* __restrict__ cur, int* __restrict__ slotT,
                            float* __restrict__ slotW) {
    const int t = blockIdx.x * blockDim.x + threadIdx.x;
    if (t >= NTOK) return;
#pragma unroll
    for (int k = 0; k < TOPK; ++k) {
        const int e = topI[t * 2 + k];
        const int pos = atomicAdd(&cur[e], 1);
        slotT[pos] = t;
        slotW[pos] = topW[t * 2 + k];
    }
}

// ---------------------------------------------------------------------------
// 4) grouped GEMM1: H[slot, f] = gelu( X[tok(slot), :] @ W1[e] + b1[e] )
//    256 threads = 8 waves; 4x2 wave grid; each wave owns a 32x64 C tile
//    (2x4 WMMA 16x16 accumulators). bf16 operands via LDS, f32 accumulate.
// ---------------------------------------------------------------------------
__global__ __launch_bounds__(256) void moe_gemm1(
    const float* __restrict__ x, const float* __restrict__ W1,
    const float* __restrict__ b1, const int* __restrict__ slotT,
    const int* __restrict__ offs, bf16_t* __restrict__ H) {
    const int e = blockIdx.z, mt = blockIdx.y, ft = blockIdx.x;
    const int base = offs[e];
    const int pcnt = offs[e + 1] - base;
    if (mt * TM >= pcnt) return;

    __shared__ __align__(16) bf16_t As[TM * LSTR];
    __shared__ __align__(16) bf16_t Bs[TN * LSTR];

    const int tid = threadIdx.x;
    const int lane = tid & 31, wid = tid >> 5;
    const int wm = wid >> 1, wn = wid & 1;      // 4x2 waves
    const int lm = lane & 15;
    const int kb = (lane >> 4) << 3;            // ISA 16-bit A/B frag k-base: 0 or 8

    v8f acc[2][4] = {};

    // staging coordinates
    const int ar = tid >> 1, ah = (tid & 1) * 16;   // A: row, k-half (16 elems)
    const int bk = tid >> 3, bf = (tid & 7) * 16;   // B: k-row, f-group (16 elems)
    const int tok = slotT[base + mt * TM + ar];
    const float* aSrc = (tok >= 0) ? (x + (size_t)tok * D_MODEL + ah) : nullptr;
    const float* bSrc = W1 + ((size_t)e * D_MODEL + bk) * D_FF + (size_t)ft * TN + bf;

    for (int kt = 0; kt < D_MODEL / TK; ++kt) {
        if (aSrc) {
#pragma unroll
            for (int j = 0; j < 16; ++j)
                As[ar * LSTR + ah + j] = (bf16_t)aSrc[kt * TK + j];
        } else {
#pragma unroll
            for (int j = 0; j < 16; ++j)
                As[ar * LSTR + ah + j] = (bf16_t)0.0f;
        }
        // stage B transposed: Bs[n][k] so B-frag load matches the A-frag pattern
#pragma unroll
        for (int j = 0; j < 16; ++j)
            Bs[(bf + j) * LSTR + bk] = (bf16_t)bSrc[(size_t)kt * TK * D_FF + j];
        __syncthreads();

        v16bf a[2], b[4];
#pragma unroll
        for (int mi = 0; mi < 2; ++mi) {
            const bf16_t* p = &As[(wm * 32 + mi * 16 + lm) * LSTR + kb];
            ((uint4*)&a[mi])[0] = *(const uint4*)(p);        // K = kb..kb+7
            ((uint4*)&a[mi])[1] = *(const uint4*)(p + 16);   // K = kb+16..kb+23
        }
#pragma unroll
        for (int ni = 0; ni < 4; ++ni) {
            const bf16_t* p = &Bs[(wn * 64 + ni * 16 + lm) * LSTR + kb];
            ((uint4*)&b[ni])[0] = *(const uint4*)(p);
            ((uint4*)&b[ni])[1] = *(const uint4*)(p + 16);
        }
#pragma unroll
        for (int mi = 0; mi < 2; ++mi)
#pragma unroll
            for (int ni = 0; ni < 4; ++ni)
                acc[mi][ni] = __builtin_amdgcn_wmma_f32_16x16x32_bf16(
                    false, a[mi], false, b[ni], (short)0, acc[mi][ni], false, false);
        __syncthreads();
    }

    // epilogue: C layout -> n = lane&15, m = v + 8*(lane>=16)
    const int mrow0 = base + mt * TM + wm * 32 + ((lane >> 4) << 3);
#pragma unroll
    for (int ni = 0; ni < 4; ++ni) {
        const int fcol = ft * TN + wn * 64 + ni * 16 + lm;
        const float bb = b1[e * D_FF + fcol];
#pragma unroll
        for (int mi = 0; mi < 2; ++mi)
#pragma unroll
            for (int v = 0; v < 8; ++v) {
                const int srow = mrow0 + mi * 16 + v;
                H[(size_t)srow * D_FF + fcol] = (bf16_t)gelu_exact(acc[mi][ni][v] + bb);
            }
    }
}

// ---------------------------------------------------------------------------
// 5) grouped GEMM2 + combine: out[tok,d] += w * (H[slot,:] @ W2[e] + b2[e])
// ---------------------------------------------------------------------------
__global__ __launch_bounds__(256) void moe_gemm2(
    const bf16_t* __restrict__ H, const float* __restrict__ W2,
    const float* __restrict__ b2, const int* __restrict__ slotT,
    const float* __restrict__ slotW, const int* __restrict__ offs,
    float* __restrict__ out) {
    const int e = blockIdx.z, mt = blockIdx.y, dt = blockIdx.x;
    const int base = offs[e];
    const int pcnt = offs[e + 1] - base;
    if (mt * TM >= pcnt) return;

    __shared__ __align__(16) bf16_t As[TM * LSTR];
    __shared__ __align__(16) bf16_t Bs[TN * LSTR];

    const int tid = threadIdx.x;
    const int lane = tid & 31, wid = tid >> 5;
    const int wm = wid >> 1, wn = wid & 1;
    const int lm = lane & 15;
    const int kb = (lane >> 4) << 3;

    v8f acc[2][4] = {};

    const int ar = tid >> 1, ah = (tid & 1) * 16;
    const int bk = tid >> 3, bf = (tid & 7) * 16;
    const bf16_t* aSrc = H + (size_t)(base + mt * TM + ar) * D_FF + ah;
    const float* bSrc = W2 + ((size_t)e * D_FF + bk) * D_MODEL + (size_t)dt * TN + bf;

    for (int kt = 0; kt < D_FF / TK; ++kt) {
#pragma unroll
        for (int j = 0; j < 16; ++j)
            As[ar * LSTR + ah + j] = aSrc[kt * TK + j];
#pragma unroll
        for (int j = 0; j < 16; ++j)
            Bs[(bf + j) * LSTR + bk] = (bf16_t)bSrc[(size_t)kt * TK * D_MODEL + j];
        __syncthreads();

        v16bf a[2], b[4];
#pragma unroll
        for (int mi = 0; mi < 2; ++mi) {
            const bf16_t* p = &As[(wm * 32 + mi * 16 + lm) * LSTR + kb];
            ((uint4*)&a[mi])[0] = *(const uint4*)(p);
            ((uint4*)&a[mi])[1] = *(const uint4*)(p + 16);
        }
#pragma unroll
        for (int ni = 0; ni < 4; ++ni) {
            const bf16_t* p = &Bs[(wn * 64 + ni * 16 + lm) * LSTR + kb];
            ((uint4*)&b[ni])[0] = *(const uint4*)(p);
            ((uint4*)&b[ni])[1] = *(const uint4*)(p + 16);
        }
#pragma unroll
        for (int mi = 0; mi < 2; ++mi)
#pragma unroll
            for (int ni = 0; ni < 4; ++ni)
                acc[mi][ni] = __builtin_amdgcn_wmma_f32_16x16x32_bf16(
                    false, a[mi], false, b[ni], (short)0, acc[mi][ni], false, false);
        __syncthreads();
    }

    const int mrow0 = base + mt * TM + wm * 32 + ((lane >> 4) << 3);
#pragma unroll
    for (int mi = 0; mi < 2; ++mi)
#pragma unroll
        for (int v = 0; v < 8; ++v) {
            const int srow = mrow0 + mi * 16 + v;
            const int t = slotT[srow];
            if (t < 0) continue;                       // padded slot
            const float w = slotW[srow];
#pragma unroll
            for (int ni = 0; ni < 4; ++ni) {
                const int dcol = dt * TN + wn * 64 + ni * 16 + lm;
                const float y = acc[mi][ni][v] + b2[e * D_MODEL + dcol];
                atomicAdd(&out[(size_t)t * D_MODEL + dcol], w * y);  // exactly 2 adds/elem
            }
        }
}

// ---------------------------------------------------------------------------
extern "C" void kernel_launch(void* const* d_in, const int* in_sizes, int n_in,
                              void* d_out, int out_size, void* d_ws, size_t ws_size,
                              hipStream_t stream) {
    (void)in_sizes; (void)n_in; (void)out_size; (void)ws_size;
    const float* x  = (const float*)d_in[0];
    const float* Wr = (const float*)d_in[1];
    const float* br = (const float*)d_in[2];
    const float* W1 = (const float*)d_in[3];
    const float* b1 = (const float*)d_in[4];
    const float* W2 = (const float*)d_in[5];
    const float* b2 = (const float*)d_in[6];
    float* out = (float*)d_out;

    char* ws = (char*)d_ws;
    int*    cnt   = (int*)(ws + OFF_CNT);
    int*    offs  = (int*)(ws + OFF_OFFS);
    int*    cur   = (int*)(ws + OFF_CUR);
    int*    topI  = (int*)(ws + OFF_TOPI);
    float*  topW  = (float*)(ws + OFF_TOPW);
    int*    slotT = (int*)(ws + OFF_SLOTT);
    float*  slotW = (float*)(ws + OFF_SLOTW);
    bf16_t* H     = (bf16_t*)(ws + OFF_H);     // ~72 MB bf16 activations

    moe_init<<<1024, 256, 0, stream>>>(out, cnt, slotT, slotW);
    moe_router<<<NTOK / 8, 256, 0, stream>>>(x, Wr, br, cnt, topI, topW);
    moe_plan<<<1, 32, 0, stream>>>(cnt, offs, cur, out + (size_t)NTOK * D_MODEL);
    moe_scatter<<<(NTOK + 255) / 256, 256, 0, stream>>>(topI, topW, cur, slotT, slotW);
    // worst case one expert holds all 4096 tokens -> 32 M-tiles; idle tiles early-exit
    moe_gemm1<<<dim3(D_FF / TN, 32, NEXP), 256, 0, stream>>>(x, W1, b1, slotT, offs, H);
    moe_gemm2<<<dim3(D_MODEL / TN, 32, NEXP), 256, 0, stream>>>(H, W2, b2, slotT, slotW, offs, out);
}
